// maskGNN_42666205118898
// MI455X (gfx1250) — matrix operand order repaired and checked by reading, not compiled, not run on hardware
//
#include <hip/hip_runtime.h>
#include <hip/hip_bf16.h>

#define NNODES 20000
#define DDIM   128
#define EEDGES 320000

typedef __bf16 bf16_t;
typedef bf16_t v16bf __attribute__((ext_vector_type(16)));
typedef float  v8f   __attribute__((ext_vector_type(8)));

union FragU { uint4 u[2]; v16bf v; };

__device__ __forceinline__ unsigned short f2bf(float f) {
  unsigned u = __float_as_uint(f);
  unsigned r = (u + 0x7FFFu + ((u >> 16) & 1u)) >> 16;
  return (unsigned short)r;
}
__device__ __forceinline__ unsigned pk2bf(float a, float b) {
  return (unsigned)f2bf(a) | ((unsigned)f2bf(b) << 16);
}
// order-preserving float->uint encoding (monotonic)
__device__ __forceinline__ unsigned encf(float f) {
  unsigned i = __float_as_uint(f);
  return (i & 0x80000000u) ? ~i : (i | 0x80000000u);
}
__device__ __forceinline__ float decf(unsigned u) {
  return (u & 0x80000000u) ? __uint_as_float(u & 0x7FFFFFFFu) : __uint_as_float(~u);
}
__device__ __forceinline__ float gelu_exact(float x) {
  return 0.5f * x * (1.0f + erff(x * 0.7071067811865476f));
}
__device__ __forceinline__ float wave_sum(float v) {
  for (int m = 16; m > 0; m >>= 1) v += __shfl_xor(v, m, 32);
  return v;
}
__device__ __forceinline__ float temporal_pe(float ev, int o) {
  int k2 = o >> 1;
  // 1/10000^(2*k2/128); ln(10000) = 9.210340371976184
  float div = expf(-(2.0f * (float)k2) * (9.210340371976184f / 128.0f));
  float pos = ev * 200.0f * div;
  return (o & 1) ? cosf(pos) : sinf(pos);
}

// ---------------------------------------------------------------------------
// Top-k mask (exact median radix-select w/ stable-argsort tie handling) and
// pack masked weights (bf16) into the wave32 WMMA B-operand layout:
//   B[k][n], tile (kt= k/32, nt = n/16): lane = (n%16) + 16*((k>>3)&1),
//   vgpr = 4*((k%32) >= 16) + (k&7)/2, half = k&1.
// transpose==1 -> B[k][n] = Wmasked[n][k]; else B[k][n] = Wmasked[k][n].
// ---------------------------------------------------------------------------
__global__ void __launch_bounds__(256) maskpack_kernel(
    const float* __restrict__ W, const float* __restrict__ S,
    int cols, int transpose,
    unsigned short* __restrict__ pack, float* __restrict__ wt_last)
{
  const int rows = DDIM;
  const int size = rows * cols;                 // 16384 or 16512
  const int j = size >> 1;                      // int((1-KEEP)*size), KEEP=0.5
  __shared__ unsigned hist[256];
  __shared__ unsigned sh_prefix;
  __shared__ int sh_target;
  __shared__ int sh_countLess;
  __shared__ unsigned keepbits[(DDIM * (DDIM + 1) + 31) / 32];  // 516 dwords

  const int tid = threadIdx.x;
  if (tid == 0) { sh_prefix = 0u; sh_target = j; }
  for (int i = tid; i < (size + 31) / 32; i += 256) keepbits[i] = 0u;
  __syncthreads();

  // 4-pass radix select: value at ascending sorted position j (0-based)
  for (int shift = 24; shift >= 0; shift -= 8) {
    for (int i = tid; i < 256; i += 256) hist[i] = 0u;
    __syncthreads();
    unsigned prefix = sh_prefix;
    unsigned maskHi = (shift == 24) ? 0u : (0xFFFFFFFFu << (shift + 8));
    for (int i = tid; i < size; i += 256) {
      unsigned e = encf(S[i]);
      if ((e & maskHi) == (prefix & maskHi))
        atomicAdd(&hist[(e >> shift) & 255u], 1u);
    }
    __syncthreads();
    if (tid == 0) {
      int tgt = sh_target;
      unsigned cum = 0; int b = 0;
      for (; b < 255; ++b) {
        unsigned h = hist[b];
        if ((int)(cum + h) > tgt) break;
        cum += h;
      }
      sh_prefix = prefix | ((unsigned)b << shift);
      sh_target = tgt - (int)cum;
    }
    __syncthreads();
  }
  const unsigned T_enc = sh_prefix;

  if (tid == 0) sh_countLess = 0;
  __syncthreads();
  {
    int lc = 0;
    for (int i = tid; i < size; i += 256) if (encf(S[i]) < T_enc) lc++;
    atomicAdd(&sh_countLess, lc);
  }
  __syncthreads();
  if (tid == 0) {
    // stable argsort: among equal values, lower flat index -> lower rank.
    // ranks >= j kept => drop first (j - countLess) equal elements.
    int need = j - sh_countLess;
    int seen = 0;
    for (int i = 0; i < size; ++i) {
      if (encf(S[i]) == T_enc) {
        if (seen >= need) keepbits[i >> 5] |= (1u << (i & 31));
        seen++;
      }
    }
  }
  __syncthreads();

  for (int idx = tid; idx < DDIM * DDIM; idx += 256) {
    int k = idx >> 7;
    int n = idx & 127;
    int wr = transpose ? n : k;
    int wc = transpose ? k : n;
    int wi = wr * cols + wc;
    unsigned e = encf(S[wi]);
    bool keep = (e > T_enc) ||
                (e == T_enc && ((keepbits[wi >> 5] >> (wi & 31)) & 1u));
    float val = keep ? W[wi] : 0.0f;
    int kt = k >> 5, kk = k & 31, nt = n >> 4, nl = n & 15;
    int lane = nl + ((kk & 8) ? 16 : 0);
    int vg = ((kk & 16) >> 2) + ((kk & 7) >> 1);
    int half = kk & 1;
    pack[(((kt * 8 + nt) * 32 + lane) * 8 + vg) * 2 + half] = f2bf(val);
  }
  if (wt_last) {
    for (int o = tid; o < DDIM; o += 256) {
      int wi = o * cols + 128;
      unsigned e = encf(S[wi]);
      bool keep = (e > T_enc) ||
                  (e == T_enc && ((keepbits[wi >> 5] >> (wi & 31)) & 1u));
      wt_last[o] = keep ? W[wi] : 0.0f;
    }
  }
}

// ---------------------------------------------------------------------------
// LayerNorm per node -> bf16 row-major [N,128]
// ---------------------------------------------------------------------------
__global__ void __launch_bounds__(256) layernorm_kernel(
    const float* __restrict__ x, const float* __restrict__ gamma,
    const float* __restrict__ beta, unsigned* __restrict__ xnb)
{
  int lane = threadIdx.x & 31;
  int n = blockIdx.x * 8 + (threadIdx.x >> 5);
  if (n >= NNODES) return;
  float4 xv = ((const float4*)(x + (size_t)n * DDIM))[lane];
  float s = wave_sum(xv.x + xv.y + xv.z + xv.w);
  float mu = s * (1.0f / DDIM);
  float dx = xv.x - mu, dy = xv.y - mu, dz = xv.z - mu, dw = xv.w - mu;
  float v = wave_sum(dx * dx + dy * dy + dz * dz + dw * dw);
  float rstd = rsqrtf(v * (1.0f / DDIM) + 1e-5f);
  float4 g = ((const float4*)gamma)[lane];
  float4 b = ((const float4*)beta)[lane];
  unsigned* row = xnb + (size_t)n * (DDIM / 2);
  row[lane * 2 + 0] = pk2bf(dx * rstd * g.x + b.x, dy * rstd * g.y + b.y);
  row[lane * 2 + 1] = pk2bf(dz * rstd * g.z + b.z, dw * rstd * g.w + b.w);
}

// ---------------------------------------------------------------------------
// Node GEMM: C[nrows,128] = A[nrows,128] @ Bpack (128x128), bf16 WMMA, f32 acc.
// One wave computes a 16x128 strip: 4 K-tiles x 8 N-tiles of
// v_wmma_f32_16x16x32_bf16. A fragment load = two b128 per K-tile (ISA A layout),
// B fragment load = two b128 per lane from the prepacked layout.
// ---------------------------------------------------------------------------
template <int A_F32, int ACC, int OUT_BF16>
__global__ void __launch_bounds__(256) gemm_node128(
    const void* __restrict__ Aptr, const unsigned* __restrict__ Bpack,
    void* __restrict__ Cptr, int nrows)
{
  int lane = threadIdx.x & 31;
  int rowTile = blockIdx.x * 8 + (threadIdx.x >> 5);
  if (rowTile * 16 >= nrows) return;
  int rowBase = rowTile * 16;
  int h = lane >> 4;        // half-wave select
  int nl = lane & 15;
  int rA = rowBase + nl;    // row this lane supplies for the A fragment

  v8f acc[8];
  if (ACC) {
    const float* Cf = (const float*)Cptr;
#pragma unroll
    for (int nt = 0; nt < 8; ++nt)
#pragma unroll
      for (int r = 0; r < 8; ++r)
        acc[nt][r] = Cf[(size_t)(rowBase + r + h * 8) * DDIM + nt * 16 + nl];
  } else {
#pragma unroll
    for (int nt = 0; nt < 8; ++nt)
#pragma unroll
      for (int r = 0; r < 8; ++r) acc[nt][r] = 0.0f;
  }

#pragma unroll
  for (int kt = 0; kt < 4; ++kt) {
    FragU af;
    if (!A_F32) {
      const uint4* Arow =
          (const uint4*)((const unsigned short*)Aptr + (size_t)rA * DDIM);
      // lanes 0-15: dwords [kt*16+0..3] & [kt*16+8..11]  (K 0-7, 16-23)
      // lanes16-31: dwords [kt*16+4..7] & [kt*16+12..15] (K 8-15, 24-31)
      af.u[0] = Arow[kt * 4 + h];
      af.u[1] = Arow[kt * 4 + 2 + h];
    } else {
      const float4* Arow =
          (const float4*)((const float*)Aptr + (size_t)rA * DDIM);
      float4 f0 = Arow[kt * 8 + 2 * h];
      float4 f1 = Arow[kt * 8 + 2 * h + 1];
      float4 f2 = Arow[kt * 8 + 4 + 2 * h];
      float4 f3 = Arow[kt * 8 + 5 + 2 * h];
      af.u[0].x = pk2bf(f0.x, f0.y); af.u[0].y = pk2bf(f0.z, f0.w);
      af.u[0].z = pk2bf(f1.x, f1.y); af.u[0].w = pk2bf(f1.z, f1.w);
      af.u[1].x = pk2bf(f2.x, f2.y); af.u[1].y = pk2bf(f2.z, f2.w);
      af.u[1].z = pk2bf(f3.x, f3.y); af.u[1].w = pk2bf(f3.z, f3.w);
    }
#pragma unroll
    for (int nt = 0; nt < 8; ++nt) {
      FragU bfr;
      const uint4* bp =
          (const uint4*)(Bpack + (size_t)(((kt * 8 + nt) * 32 + lane) * 8));
      bfr.u[0] = bp[0];
      bfr.u[1] = bp[1];
      acc[nt] = __builtin_amdgcn_wmma_f32_16x16x32_bf16(
          false, af.v, false, bfr.v, (short)0, acc[nt], false, false);
    }
  }

#pragma unroll
  for (int nt = 0; nt < 8; ++nt)
#pragma unroll
    for (int r = 0; r < 8; ++r) {
      size_t o = (size_t)(rowBase + r + h * 8) * DDIM + nt * 16 + nl;
      if (OUT_BF16) ((unsigned short*)Cptr)[o] = f2bf(acc[nt][r]);
      else          ((float*)Cptr)[o] = acc[nt][r];
    }
}

// ---------------------------------------------------------------------------
// Edge phase 1: attn[e] = xjt(e) . G{s,d}[dst] / sqrt(128); segment max (enc).
// One wave per edge, 4 dims per lane.
// ---------------------------------------------------------------------------
__global__ void __launch_bounds__(256) edge_attn_kernel(
    const int* __restrict__ ei, const float* __restrict__ ev_arr,
    const float* __restrict__ es_arr, const float* __restrict__ Tn,
    const float* __restrict__ wt_last, const float* __restrict__ Gs,
    const float* __restrict__ Gd, float* __restrict__ attn,
    unsigned* __restrict__ m_u)
{
  int lane = threadIdx.x & 31;
  int e = blockIdx.x * 8 + (threadIdx.x >> 5);
  if (e >= EEDGES) return;
  int src = ei[e], dst = ei[EEDGES + e];
  float ev = ev_arr[e];
  bool same = es_arr[e] != 0.0f;
  float4 t = ((const float4*)(Tn + (size_t)src * DDIM))[lane];
  float4 wl = ((const float4*)wt_last)[lane];
  int o0 = lane * 4;
  float xj0 = gelu_exact(t.x + ev * wl.x) + temporal_pe(ev, o0 + 0);
  float xj1 = gelu_exact(t.y + ev * wl.y) + temporal_pe(ev, o0 + 1);
  float xj2 = gelu_exact(t.z + ev * wl.z) + temporal_pe(ev, o0 + 2);
  float xj3 = gelu_exact(t.w + ev * wl.w) + temporal_pe(ev, o0 + 3);
  const float* G = same ? Gs : Gd;
  float4 g = ((const float4*)(G + (size_t)dst * DDIM))[lane];
  float tot = wave_sum(xj0 * g.x + xj1 * g.y + xj2 * g.z + xj3 * g.w);
  if (lane == 0) {
    float a = tot * 0.08838834764831843f;  // 1/sqrt(128)
    attn[e] = a;
    atomicMax(&m_u[dst], encf(a));
  }
}

// Edge phase 2: a = exp(attn - m[dst]); segment sum.
__global__ void __launch_bounds__(256) edge_exp_kernel(
    const int* __restrict__ ei, const float* __restrict__ attn,
    const unsigned* __restrict__ m_u, float* __restrict__ abuf,
    float* __restrict__ ssum)
{
  int e = blockIdx.x * 256 + threadIdx.x;
  if (e >= EEDGES) return;
  int dst = ei[EEDGES + e];
  float a = expf(attn[e] - decf(m_u[dst]));
  abuf[e] = a;
  atomicAdd(&ssum[dst], a);
}

// Edge phase 3: u_{s,d}[dst] += attn_n * xjt(e)   (recompute xjt; L2-resident)
__global__ void __launch_bounds__(256) edge_scatter_kernel(
    const int* __restrict__ ei, const float* __restrict__ ev_arr,
    const float* __restrict__ es_arr, const float* __restrict__ Tn,
    const float* __restrict__ wt_last, const float* __restrict__ abuf,
    const float* __restrict__ ssum, float* __restrict__ u_s,
    float* __restrict__ u_d)
{
  int lane = threadIdx.x & 31;
  int e = blockIdx.x * 8 + (threadIdx.x >> 5);
  if (e >= EEDGES) return;
  int src = ei[e], dst = ei[EEDGES + e];
  float ev = ev_arr[e];
  bool same = es_arr[e] != 0.0f;
  float4 t = ((const float4*)(Tn + (size_t)src * DDIM))[lane];
  float4 wl = ((const float4*)wt_last)[lane];
  int o0 = lane * 4;
  float xj0 = gelu_exact(t.x + ev * wl.x) + temporal_pe(ev, o0 + 0);
  float xj1 = gelu_exact(t.y + ev * wl.y) + temporal_pe(ev, o0 + 1);
  float xj2 = gelu_exact(t.z + ev * wl.z) + temporal_pe(ev, o0 + 2);
  float xj3 = gelu_exact(t.w + ev * wl.w) + temporal_pe(ev, o0 + 3);
  float w = abuf[e] / (ssum[dst] + 1e-16f);
  float* U = (same ? u_s : u_d) + (size_t)dst * DDIM + o0;
  atomicAdd(U + 0, w * xj0);
  atomicAdd(U + 1, w * xj1);
  atomicAdd(U + 2, w * xj2);
  atomicAdd(U + 3, w * xj3);
}

// Final: out = x + gelu(aggr)
__global__ void __launch_bounds__(256) final_kernel(
    const float* __restrict__ x, const float* __restrict__ aggr,
    float* __restrict__ out)
{
  int i = blockIdx.x * 256 + threadIdx.x;
  if (i >= NNODES * DDIM) return;
  out[i] = x[i] + gelu_exact(aggr[i]);
}

// ---------------------------------------------------------------------------
extern "C" void kernel_launch(void* const* d_in, const int* in_sizes, int n_in,
                              void* d_out, int out_size, void* d_ws,
                              size_t ws_size, hipStream_t stream) {
  const float* x     = (const float*)d_in[0];
  const int*   ei    = (const int*)d_in[1];
  const float* ev    = (const float*)d_in[2];
  const float* esame = (const float*)d_in[3];
  const float* gam   = (const float*)d_in[4];
  const float* bet   = (const float*)d_in[5];
  const float* Wt  = (const float*)d_in[6];   const float* St  = (const float*)d_in[7];
  const float* Wks = (const float*)d_in[8];   const float* Sks = (const float*)d_in[9];
  const float* Wkd = (const float*)d_in[10];  const float* Skd = (const float*)d_in[11];
  const float* Wq  = (const float*)d_in[12];  const float* Sq  = (const float*)d_in[13];
  const float* Wvs = (const float*)d_in[14];  const float* Svs = (const float*)d_in[15];
  const float* Wvd = (const float*)d_in[16];  const float* Svd = (const float*)d_in[17];
  float* out = (float*)d_out;

  // workspace carve-up (256B aligned)
  char* ws = (char*)d_ws;
  size_t off = 0;
  auto alloc = [&](size_t bytes) {
    char* p = ws + off;
    off = (off + bytes + 255) & ~(size_t)255;
    return p;
  };
  unsigned*       xnb   = (unsigned*)alloc((size_t)NNODES * DDIM * 2);
  float*          Tn    = (float*)alloc((size_t)NNODES * DDIM * 4);
  unsigned short* Qb    = (unsigned short*)alloc((size_t)NNODES * DDIM * 2);
  float*          Gsb   = (float*)alloc((size_t)NNODES * DDIM * 4);
  float*          Gdb   = (float*)alloc((size_t)NNODES * DDIM * 4);
  float*          u_s   = (float*)alloc((size_t)NNODES * DDIM * 4);
  float*          u_d   = (float*)alloc((size_t)NNODES * DDIM * 4);
  float*          aggr  = (float*)alloc((size_t)NNODES * DDIM * 4);
  float*          attn  = (float*)alloc((size_t)EEDGES * 4);
  float*          abuf  = (float*)alloc((size_t)EEDGES * 4);
  unsigned*       m_u   = (unsigned*)alloc((size_t)NNODES * 4);
  float*          ssum  = (float*)alloc((size_t)NNODES * 4);
  unsigned short* pWtT  = (unsigned short*)alloc((size_t)DDIM * DDIM * 2);
  unsigned short* pWqT  = (unsigned short*)alloc((size_t)DDIM * DDIM * 2);
  unsigned short* pWks  = (unsigned short*)alloc((size_t)DDIM * DDIM * 2);
  unsigned short* pWkd  = (unsigned short*)alloc((size_t)DDIM * DDIM * 2);
  unsigned short* pWvsT = (unsigned short*)alloc((size_t)DDIM * DDIM * 2);
  unsigned short* pWvdT = (unsigned short*)alloc((size_t)DDIM * DDIM * 2);
  float*          wlast = (float*)alloc((size_t)DDIM * 4);

  hipMemsetAsync(u_s, 0, (size_t)NNODES * DDIM * 4, stream);
  hipMemsetAsync(u_d, 0, (size_t)NNODES * DDIM * 4, stream);
  hipMemsetAsync(ssum, 0, (size_t)NNODES * 4, stream);
  hipMemsetAsync(m_u, 0, (size_t)NNODES * 4, stream);  // enc-space -inf

  // masked + packed weights
  maskpack_kernel<<<1, 256, 0, stream>>>(Wt,  St,  DDIM + 1, 1, pWtT,  wlast);
  maskpack_kernel<<<1, 256, 0, stream>>>(Wq,  Sq,  DDIM,     1, pWqT,  nullptr);
  maskpack_kernel<<<1, 256, 0, stream>>>(Wks, Sks, DDIM,     0, pWks,  nullptr);
  maskpack_kernel<<<1, 256, 0, stream>>>(Wkd, Skd, DDIM,     0, pWkd,  nullptr);
  maskpack_kernel<<<1, 256, 0, stream>>>(Wvs, Svs, DDIM,     1, pWvsT, nullptr);
  maskpack_kernel<<<1, 256, 0, stream>>>(Wvd, Svd, DDIM,     1, pWvdT, nullptr);

  layernorm_kernel<<<NNODES / 8, 256, 0, stream>>>(x, gam, bet, xnb);

  const int gemmGrid = (NNODES / 16 + 7) / 8;  // 1250 row tiles, 8 waves/block
  // T = xn @ Wt'^T ; Q = xn @ Wq^T (bf16) ; Gs = Q @ Wks ; Gd = Q @ Wkd
  gemm_node128<0, 0, 0><<<gemmGrid, 256, 0, stream>>>(xnb, (unsigned*)pWtT, Tn, NNODES);
  gemm_node128<0, 0, 1><<<gemmGrid, 256, 0, stream>>>(xnb, (unsigned*)pWqT, Qb, NNODES);
  gemm_node128<0, 0, 0><<<gemmGrid, 256, 0, stream>>>(Qb, (unsigned*)pWks, Gsb, NNODES);
  gemm_node128<0, 0, 0><<<gemmGrid, 256, 0, stream>>>(Qb, (unsigned*)pWkd, Gdb, NNODES);

  // edge phases
  edge_attn_kernel<<<EEDGES / 8, 256, 0, stream>>>(ei, ev, esame, Tn, wlast,
                                                   Gsb, Gdb, attn, m_u);
  edge_exp_kernel<<<EEDGES / 256, 256, 0, stream>>>(ei, attn, m_u, abuf, ssum);
  edge_scatter_kernel<<<EEDGES / 8, 256, 0, stream>>>(ei, ev, esame, Tn, wlast,
                                                      abuf, ssum, u_s, u_d);

  // aggr = u_s @ Wvs^T + u_d @ Wvd^T
  gemm_node128<1, 0, 0><<<gemmGrid, 256, 0, stream>>>(u_s, (unsigned*)pWvsT, aggr, NNODES);
  gemm_node128<1, 1, 0><<<gemmGrid, 256, 0, stream>>>(u_d, (unsigned*)pWvdT, aggr, NNODES);

  final_kernel<<<(NNODES * DDIM) / 256, 256, 0, stream>>>(x, aggr, out);
}
